// TGAN_19404662243847
// MI455X (gfx1250) — matrix operand order repaired and checked by reading, not compiled, not run on hardware
//
#include <hip/hip_runtime.h>
#include <hip/hip_bf16.h>

typedef _Float16 half_t;
typedef _Float16 v16h __attribute__((ext_vector_type(16)));
typedef float    v8f  __attribute__((ext_vector_type(8)));

#define B_SZ  4096
#define N_SZ  64
#define D_SZ  128
#define M_SZ  384
#define DK_SZ 192

// ---------------------------------------------------------------------------
// u[h][m] = sum_d w_map[DK+d] * Wk[h*DK+d][m]   (k-side logit folds to a GEMV)
// ---------------------------------------------------------------------------
__global__ void k_prep_u(const float* __restrict__ Wk,
                         const float* __restrict__ w_map,
                         float* __restrict__ u) {
    int o = blockIdx.x * blockDim.x + threadIdx.x;
    if (o >= 2 * M_SZ) return;
    int h = o / M_SZ, m = o % M_SZ;
    const float* wrow = Wk + (size_t)(h * DK_SZ) * M_SZ + m;
    float acc = 0.f;
    for (int d = 0; d < DK_SZ; ++d)
        acc += w_map[DK_SZ + d] * wrow[(size_t)d * M_SZ];
    u[o] = acc;
}

__global__ void k_cvt_f16(const float* __restrict__ src,
                          half_t* __restrict__ dst, int n) {
    int i = blockIdx.x * blockDim.x + threadIdx.x;
    if (i < n) dst[i] = (half_t)src[i];
}

// ---------------------------------------------------------------------------
// Streaming attention kernel: one workgroup per batch element.
//  - streams k_in = [seq | seq_e | seq_t]  (the only pass over 384 MB)
//  - fused logit dot with u (registers), masked softmax
//  - WMMA: c(2x384) = attn_pad(16x64) @ k_in(64x384); both operands staged as
//    f16 in LDS with conflict-free 72-half pitch so fragments load as b128.
// ---------------------------------------------------------------------------
__global__ __launch_bounds__(256) void k_attn(
    const float* __restrict__ seq, const float* __restrict__ seq_e,
    const float* __restrict__ seq_t, const unsigned char* __restrict__ mask,
    const float* __restrict__ u_g, float* __restrict__ attn_out,
    half_t* __restrict__ c16)
{
    constexpr int KP = 72;                    // pitch in halves: 144B rows, 16B aligned,
    constexpr int AP = 72;                    // bank stride 36 -> conflict-free over 16 rows
    __shared__ half_t kinT[M_SZ * KP];        // [feature][seq]   54 KB
    __shared__ half_t attn16[16 * AP];        // padded 16x64 A-matrix (f16)
    __shared__ float  u_lds[2 * M_SZ];
    __shared__ float  s_lds[2][N_SZ];
    __shared__ float  attn_lds[2][N_SZ];

    const int b   = blockIdx.x;
    const int tid = threadIdx.x;

    for (int i = tid; i < 2 * M_SZ; i += 256) u_lds[i] = u_g[i];
    for (int i = tid; i < 16 * AP; i += 256) attn16[i] = (half_t)0.f;

    // ---- streaming load + fused logit dot: 4 threads per sequence row ----
    const int n = tid >> 2;                   // 0..63
    const int q = tid & 3;                    // 96-feature span each
    const float* p0 = seq   + (size_t)(b * N_SZ + n) * D_SZ;
    const float* p1 = seq_e + (size_t)(b * N_SZ + n) * D_SZ;
    const float* p2 = seq_t + (size_t)(b * N_SZ + n) * D_SZ;
    float acc0 = 0.f, acc1 = 0.f;
    const int base = q * 96;
#pragma unroll
    for (int i = 0; i < 24; ++i) {
        int mm = base + i * 4;
        int s  = mm >> 7;
        const float* p = (s == 0) ? p0 : (s == 1) ? p1 : p2;
        float4 v = *(const float4*)(p + (mm & 127));
        acc0 += v.x * u_lds[mm] + v.y * u_lds[mm + 1] +
                v.z * u_lds[mm + 2] + v.w * u_lds[mm + 3];
        acc1 += v.x * u_lds[M_SZ + mm] + v.y * u_lds[M_SZ + mm + 1] +
                v.z * u_lds[M_SZ + mm + 2] + v.w * u_lds[M_SZ + mm + 3];
        kinT[(mm + 0) * KP + n] = (half_t)v.x;
        kinT[(mm + 1) * KP + n] = (half_t)v.y;
        kinT[(mm + 2) * KP + n] = (half_t)v.z;
        kinT[(mm + 3) * KP + n] = (half_t)v.w;
    }
    acc0 += __shfl_xor(acc0, 1); acc0 += __shfl_xor(acc0, 2);
    acc1 += __shfl_xor(acc1, 1); acc1 += __shfl_xor(acc1, 2);
    if (q == 0) {
        bool mk = mask[b * N_SZ + n] != 0;
        s_lds[0][n] = mk ? -1e10f : acc0;     // q-side logit term cancels in softmax
        s_lds[1][n] = mk ? -1e10f : acc1;
    }
    __syncthreads();

    // ---- masked softmax over N=64 (tiny, serial per head) ----
    if (tid < 2) {
        float mx = -1e30f;
        for (int i = 0; i < N_SZ; ++i) mx = fmaxf(mx, s_lds[tid][i]);
        float ssum = 0.f;
        for (int i = 0; i < N_SZ; ++i) {
            float e = __expf(s_lds[tid][i] - mx);
            attn_lds[tid][i] = e; ssum += e;
        }
        float inv = 1.f / ssum;
        for (int i = 0; i < N_SZ; ++i) attn_lds[tid][i] *= inv;
    }
    __syncthreads();

    if (tid < 128) {                          // write attn output + padded f16 tile
        int h = tid >> 6, i = tid & 63;
        float a = attn_lds[h][i];
        attn_out[((size_t)h * B_SZ + b) * N_SZ + i] = a;
        attn16[h * AP + i] = (half_t)a;
    }
    __syncthreads();

    // ---- c = attn_pad(16x64) @ k_in(64x384) via v_wmma_f32_16x16x32_f16 ----
    const int lane = tid & 31, wv = tid >> 5;
    const int hi = lane >> 4, lm16 = lane & 15;
    v8f d0 = {0,0,0,0,0,0,0,0}, d1 = d0, d2 = d0;
#pragma unroll
    for (int ks = 0; ks < 2; ++ks) {
        // A fragment: row m = lane%16, K chunks [hi*8..+7] and [16+hi*8..+7]
        union { v16h v; uint4 u[2]; } a;
        a.u[0] = *(const uint4*)&attn16[lm16 * AP + ks * 32 + hi * 8];
        a.u[1] = *(const uint4*)&attn16[lm16 * AP + ks * 32 + 16 + hi * 8];
#pragma unroll
        for (int t = 0; t < 3; ++t) {
            int ft = wv * 3 + t;              // 24 feature tiles over 8 waves
            union { v16h v; uint4 u[2]; } bf; // B-matrix 32x16: k chunks [hi*8..+7],[16+hi*8..+7], n = lane%16
            const half_t* pb =
                &kinT[(ft * 16 + lm16) * KP + ks * 32 + hi * 8];
            bf.u[0] = *(const uint4*)pb; bf.u[1] = *(const uint4*)(pb + 16);
            v8f& d = (t == 0) ? d0 : (t == 1) ? d1 : d2;
            d = __builtin_amdgcn_wmma_f32_16x16x32_f16(
                    false, a.v, false, bf.v, (short)0, d, false, false);
        }
    }
    if (hi == 0) {                            // D rows 0,1 = heads 0,1 (VGPR0/1, lanes 0-15)
#pragma unroll
        for (int t = 0; t < 3; ++t) {
            int ft = wv * 3 + t;
            v8f& d = (t == 0) ? d0 : (t == 1) ? d1 : d2;
            c16[(size_t)b * 768 +         ft * 16 + lm16] = (half_t)d[0];
            c16[(size_t)b * 768 + M_SZ +  ft * 16 + lm16] = (half_t)d[1];
        }
    }
}

// ---------------------------------------------------------------------------
// Generic tiled WMMA GEMM:  Out[r][c] = act( A[r][:K] . W[c][:K] + bias[c] )
// 64x64 tile per WG (8 waves, 2 accumulators each). K-tiles are brought into
// LDS with double-buffered GLOBAL_LOAD_ASYNC_TO_LDS_B128 (ASYNCcnt DMA path,
// no VGPR round-trip), overlapping the next tile's fetch with current WMMAs.
// Requires K%32==0, ROWS%64==0, OUT%64==0 (all shapes here satisfy this).
// ---------------------------------------------------------------------------
__global__ __launch_bounds__(256) void k_gemm(
    const half_t* __restrict__ A, int lda,
    const half_t* __restrict__ W, int ldw,
    const float* __restrict__ bias,
    float* __restrict__ outF, half_t* __restrict__ outH, int ldo,
    int K, int applyRelu)
{
    constexpr int LP = 40;                    // pitch: 80B rows -> conflict-free frags
    __shared__ half_t As[2][64 * LP];
    __shared__ half_t Ws[2][64 * LP];

    const int tid = threadIdx.x;
    const int lane = tid & 31, wv = tid >> 5;
    const int hi = lane >> 4, lm = lane & 15;
    const int rowBase = blockIdx.x * 64, colBase = blockIdx.y * 64;
    const int tc = wv & 3, tr0 = (wv >> 2) * 2;
    const int lr = tid >> 2, lq = tid & 3;

    const half_t* gA = A + (size_t)(rowBase + lr) * lda + lq * 8;
    const half_t* gW = W + (size_t)(colBase + lr) * ldw + lq * 8;
    // LDS byte addresses for this thread's 16B staging slots (flat addr low 32
    // bits == LDS offset, ISA 10.2) — one slot per buffer.
    unsigned ldsA0 = (unsigned)(size_t)&As[0][lr * LP + lq * 8];
    unsigned ldsA1 = (unsigned)(size_t)&As[1][lr * LP + lq * 8];
    unsigned ldsW0 = (unsigned)(size_t)&Ws[0][lr * LP + lq * 8];
    unsigned ldsW1 = (unsigned)(size_t)&Ws[1][lr * LP + lq * 8];

    v8f acc0 = {0,0,0,0,0,0,0,0}, acc1 = acc0;

    const int nstages = K >> 5;
    auto issue = [&](int stage, int buf) {
        unsigned long long pa = (unsigned long long)(gA + stage * 32);
        unsigned long long pw = (unsigned long long)(gW + stage * 32);
        unsigned la = buf ? ldsA1 : ldsA0;
        unsigned lw = buf ? ldsW1 : ldsW0;
        asm volatile("global_load_async_to_lds_b128 %0, %1, off\n\t"
                     "global_load_async_to_lds_b128 %2, %3, off"
                     :: "v"(la), "v"(pa), "v"(lw), "v"(pw) : "memory");
    };

    issue(0, 0);
    for (int s = 0; s < nstages; ++s) {
        const int buf = s & 1;
        if (s + 1 < nstages) {
            issue(s + 1, buf ^ 1);
            asm volatile("s_wait_asynccnt 0x2" ::: "memory"); // stage s retired
        } else {
            asm volatile("s_wait_asynccnt 0x0" ::: "memory");
        }
        __syncthreads();

        const half_t* Ab = As[buf];
        const half_t* Wb = Ws[buf];
        union { v16h v; uint4 u[2]; } bf, a0, a1;
        const half_t* pb = &Wb[(tc * 16 + lm) * LP + hi * 8];
        bf.u[0] = *(const uint4*)pb; bf.u[1] = *(const uint4*)(pb + 16);
        const int r0 = (tr0 + 0) * 16 + lm, r1 = (tr0 + 1) * 16 + lm;
        a0.u[0] = *(const uint4*)&Ab[r0 * LP + hi * 8];
        a0.u[1] = *(const uint4*)&Ab[r0 * LP + 16 + hi * 8];
        a1.u[0] = *(const uint4*)&Ab[r1 * LP + hi * 8];
        a1.u[1] = *(const uint4*)&Ab[r1 * LP + 16 + hi * 8];

        acc0 = __builtin_amdgcn_wmma_f32_16x16x32_f16(
                   false, a0.v, false, bf.v, (short)0, acc0, false, false);
        acc1 = __builtin_amdgcn_wmma_f32_16x16x32_f16(
                   false, a1.v, false, bf.v, (short)0, acc1, false, false);
        __syncthreads();
    }

    const int col = colBase + tc * 16 + lm;
    const float bv = bias ? bias[col] : 0.f;
#pragma unroll
    for (int i = 0; i < 8; ++i) {
        int m = i + hi * 8;
        float v0 = acc0[i] + bv, v1 = acc1[i] + bv;
        if (applyRelu) { v0 = fmaxf(v0, 0.f); v1 = fmaxf(v1, 0.f); }
        size_t r0 = (size_t)(rowBase + (tr0 + 0) * 16 + m) * ldo + col;
        size_t r1 = (size_t)(rowBase + (tr0 + 1) * 16 + m) * ldo + col;
        if (outF) { outF[r0] = v0; outF[r1] = v1; }
        else      { outH[r0] = (half_t)v0; outH[r1] = (half_t)v1; }
    }
}

// ---------------------------------------------------------------------------
// LeakyReLU + residual(q_in) + LayerNorm + build x = [y | src] (f16).
// One row per wave (8 rows / WG), shuffle reductions.
// ---------------------------------------------------------------------------
__global__ __launch_bounds__(256) void k_ln(
    const float* __restrict__ tmp, const float* __restrict__ src,
    const float* __restrict__ src_s, const float* __restrict__ src_t,
    const float* __restrict__ ln_g, const float* __restrict__ ln_b,
    half_t* __restrict__ x16)
{
    const int lane = threadIdx.x & 31, wv = threadIdx.x >> 5;
    const int b = blockIdx.x * 8 + wv;
    float v[12], s1 = 0.f, s2 = 0.f;
#pragma unroll
    for (int j = 0; j < 12; ++j) {
        int e = lane + 32 * j;
        float t = tmp[(size_t)b * M_SZ + e];
        t = (t > 0.f) ? t : 0.2f * t;                         // leaky_relu
        float r = (e < 128) ? src[(size_t)b * D_SZ + e]
                : (e < 256) ? src_s[(size_t)b * D_SZ + e - 128]
                            : src_t[(size_t)b * D_SZ + e - 256];
        t += r;
        v[j] = t; s1 += t; s2 += t * t;
    }
#pragma unroll
    for (int o = 16; o; o >>= 1) { s1 += __shfl_xor(s1, o); s2 += __shfl_xor(s2, o); }
    float mu  = s1 * (1.f / M_SZ);
    float var = s2 * (1.f / M_SZ) - mu * mu;
    float inv = rsqrtf(var + 1e-5f);
#pragma unroll
    for (int j = 0; j < 12; ++j) {
        int e = lane + 32 * j;
        x16[(size_t)b * 512 + e] = (half_t)((v[j] - mu) * inv * ln_g[e] + ln_b[e]);
    }
#pragma unroll
    for (int j = 0; j < 4; ++j) {
        int e = lane + 32 * j;
        x16[(size_t)b * 512 + M_SZ + e] = (half_t)src[(size_t)b * D_SZ + e];
    }
}

// ---------------------------------------------------------------------------
extern "C" void kernel_launch(void* const* d_in, const int* in_sizes, int n_in,
                              void* d_out, int out_size, void* d_ws, size_t ws_size,
                              hipStream_t stream) {
    const float* src   = (const float*)d_in[0];
    const float* src_t = (const float*)d_in[1];
    const float* src_s = (const float*)d_in[2];
    const float* seq   = (const float*)d_in[3];
    const float* seq_t = (const float*)d_in[4];
    const float* seq_e = (const float*)d_in[5];
    const unsigned char* mask = (const unsigned char*)d_in[6];
    // d_in[7] = Wq (dead: q-side softmax term is constant along N and cancels)
    const float* Wk    = (const float*)d_in[8];
    const float* Wv    = (const float*)d_in[9];
    const float* w_map = (const float*)d_in[10];
    const float* Wfc   = (const float*)d_in[11];
    const float* bfc   = (const float*)d_in[12];
    const float* ln_g  = (const float*)d_in[13];
    const float* ln_b  = (const float*)d_in[14];
    const float* W1    = (const float*)d_in[15];
    const float* b1    = (const float*)d_in[16];
    const float* W2    = (const float*)d_in[17];
    const float* b2    = (const float*)d_in[18];
    float* out = (float*)d_out;                 // [B*D output | H*B*N attn]

    char* ws = (char*)d_ws;
    size_t off = 0;
    auto wsAlloc = [&](size_t bytes) -> void* {
        void* p = (void*)(ws + off);
        off += (bytes + 255) & ~(size_t)255;
        return p;
    };
    float*  u      = (float*)wsAlloc(2 * M_SZ * sizeof(float));
    half_t* Wv16   = (half_t*)wsAlloc((size_t)M_SZ * M_SZ * 2);
    half_t* Wfc16  = (half_t*)wsAlloc((size_t)M_SZ * M_SZ * 2);
    half_t* W1_16  = (half_t*)wsAlloc((size_t)D_SZ * 512 * 2);
    half_t* W2_16  = (half_t*)wsAlloc((size_t)D_SZ * D_SZ * 2);
    half_t* c16    = (half_t*)wsAlloc((size_t)B_SZ * 768 * 2);
    half_t* vout16 = (half_t*)wsAlloc((size_t)B_SZ * M_SZ * 2);
    float*  tmpf   = (float*)wsAlloc((size_t)B_SZ * M_SZ * 4);
    half_t* x16    = (half_t*)wsAlloc((size_t)B_SZ * 512 * 2);
    half_t* h16    = (half_t*)wsAlloc((size_t)B_SZ * D_SZ * 2);

    // prep: u vector + f16 weight copies (resident in L2 thereafter)
    k_prep_u<<<3, 256, 0, stream>>>(Wk, w_map, u);
    k_cvt_f16<<<(M_SZ * M_SZ + 255) / 256, 256, 0, stream>>>(Wv,  Wv16,  M_SZ * M_SZ);
    k_cvt_f16<<<(M_SZ * M_SZ + 255) / 256, 256, 0, stream>>>(Wfc, Wfc16, M_SZ * M_SZ);
    k_cvt_f16<<<(D_SZ * 512  + 255) / 256, 256, 0, stream>>>(W1,  W1_16, D_SZ * 512);
    k_cvt_f16<<<(D_SZ * D_SZ + 255) / 256, 256, 0, stream>>>(W2,  W2_16, D_SZ * D_SZ);

    // the single HBM-bound pass: logits, softmax, attn_out, context c
    k_attn<<<B_SZ, 256, 0, stream>>>(seq, seq_e, seq_t, mask, u,
                                     out + (size_t)B_SZ * D_SZ, c16);

    // out0[:, h*192:(h+1)*192] = c_h @ Wv_h^T    (two 4096x192 GEMMs, K=384)
    for (int h = 0; h < 2; ++h) {
        k_gemm<<<dim3(B_SZ / 64, DK_SZ / 64), 256, 0, stream>>>(
            c16 + h * M_SZ, 768, Wv16 + (size_t)h * DK_SZ * M_SZ, M_SZ,
            nullptr, nullptr, vout16 + h * DK_SZ, M_SZ, M_SZ, 0);
    }
    // tmp = vout @ Wfc^T + bfc   (4096x384, K=384)
    k_gemm<<<dim3(B_SZ / 64, M_SZ / 64), 256, 0, stream>>>(
        vout16, M_SZ, Wfc16, M_SZ, bfc, tmpf, nullptr, M_SZ, M_SZ, 0);
    // leaky_relu + residual + LN + concat src -> x16 (B x 512)
    k_ln<<<B_SZ / 8, 256, 0, stream>>>(tmpf, src, src_s, src_t, ln_g, ln_b, x16);
    // h = relu(x @ W1^T + b1)    (4096x128, K=512)
    k_gemm<<<dim3(B_SZ / 64, D_SZ / 64), 256, 0, stream>>>(
        x16, 512, W1_16, 512, b1, nullptr, h16, D_SZ, 512, 1);
    // output = h @ W2^T + b2     (4096x128, K=128) -> d_out[0 : B*D)
    k_gemm<<<dim3(B_SZ / 64, D_SZ / 64), 256, 0, stream>>>(
        h16, D_SZ, W2_16, D_SZ, b2, out, nullptr, D_SZ, D_SZ, 0);
}
